// FastImageMamba_29257317220510
// MI455X (gfx1250) — matrix operand order repaired and checked by reading, not compiled, not run on hardware
//
#include <hip/hip_runtime.h>
#include <hip/hip_bf16.h>

// FastImageMamba for MI455X (gfx1250, wave32).
// Dense linear algebra (conv2 via im2col + the 4 Mamba projections) runs on a
// WMMA f16->f32 tiled GEMM: 64x64 block tile, 8 wave32 (2Mx4N), 2 accumulators
// per wave -> 2 v_wmma_f32_16x16x32_f16 per wave per K-step, vectorized
// staging (b128 global loads, b64 LDS stores) and global_prefetch of the next
// K tile. Everything fits in the 192MB L2; serial cost is the L=1024 scan.
// Workspace requirement: ~67 MB (stem im2col region overlays Mamba scratch).

typedef __attribute__((ext_vector_type(16))) _Float16 v16h;
typedef __attribute__((ext_vector_type(8)))  float    v8f;
typedef __attribute__((ext_vector_type(4)))  _Float16 h4;
typedef __attribute__((ext_vector_type(4)))  float    f32x4;

// ---- model constants -------------------------------------------------------
#define MB    4          // batch
#define DM    256        // d_model
#define NL    4          // layers
#define DI    512        // d_inner
#define NS    16         // d_state
#define DTR   32         // dt_rank
#define LSEQ  1024       // sequence length (32*32)
#define NTOK  (MB*LSEQ)  // 4096 tokens
#define NC    1000
#define EPSF  1e-5f

// ---- workspace layout (bytes) ---------------------------------------------
static constexpr size_t OFF_S     = 0;                                 // [4096,256] f32
static constexpr size_t STAGE     = OFF_S + (size_t)NTOK*DM*4;
// stage 1 (conv stem) -- dead once sequence s is produced
static constexpr size_t OFF_H1    = STAGE;                             // [4,64,64,128] f32 NHWC
static constexpr size_t OFF_H2    = OFF_H1 + 4ull*64*64*128*4;         // [4,64,64,256] f32 NHWC
static constexpr size_t OFF_COL   = OFF_H2 + 4ull*64*64*256*4;         // [16384,1152] f16
// stage 2 (mamba loop) -- overlays stage 1
static constexpr size_t OFF_XR    = STAGE;                             // [4096,1024]
static constexpr size_t OFF_XC    = OFF_XR    + (size_t)NTOK*2*DI*4;   // [4096,512]
static constexpr size_t OFF_XDBL  = OFF_XC    + (size_t)NTOK*DI*4;     // [4096,64]
static constexpr size_t OFF_DELTA = OFF_XDBL  + (size_t)NTOK*64*4;     // [4096,512]
static constexpr size_t OFF_Y     = OFF_DELTA + (size_t)NTOK*DI*4;     // [4096,512]
static constexpr size_t OFF_OUT   = OFF_Y     + (size_t)NTOK*DI*4;     // [4096,256]
static constexpr size_t OFF_POOL  = OFF_OUT   + (size_t)NTOK*DM*4;     // [4,256]
static constexpr size_t OFF_POOLN = OFF_POOL  + 4096;                  // [4,256]

// ---- small device helpers --------------------------------------------------
__device__ __forceinline__ float gelu_f(float x) {
  return 0.5f * x * (1.0f + erff(x * 0.70710678118654752f));
}
__device__ __forceinline__ float softplus_f(float x) {
  return (x > 20.0f) ? x : log1pf(expf(x));
}
__device__ __forceinline__ float silu_f(float x) {
  return x / (1.0f + expf(-x));
}
__device__ __forceinline__ float block_reduce_sum256(float v, float* red) {
  int tid = threadIdx.x;
  red[tid] = v; __syncthreads();
  for (int s = 128; s > 0; s >>= 1) {
    if (tid < s) red[tid] += red[tid + s];
    __syncthreads();
  }
  float r = red[0]; __syncthreads();
  return r;
}

// 4-element loaders: f32x4 -> h4 convert, or straight f16x4 copy (im2col path)
__device__ __forceinline__ h4 load4_h(const float* p) {
  f32x4 v = *(const f32x4*)p;
  h4 r;
  r[0] = (_Float16)v[0]; r[1] = (_Float16)v[1];
  r[2] = (_Float16)v[2]; r[3] = (_Float16)v[3];
  return r;
}
__device__ __forceinline__ h4 load4_h(const _Float16* p) {
  return *(const h4*)p;
}

// ---- WMMA GEMM:  C[M,N] = act(A[M,K] @ W[N,K]^T + bias[N]) -----------------
// 256 threads = 8 wave32, block tile 64(M) x 64(N), K step 32.
// Wave grid 2(M) x 4(N); each wave owns two 16x16 D tiles stacked in M
// (B fragment reused) -> 2 v_wmma_f32_16x16x32_f16 per wave per K-step.
// Staging: 4 consecutive K elems per thread (global b128/b64 -> ds b64),
// plus global_prefetch_b8 of the next K tile.
template <typename TA>
__global__ __launch_bounds__(256) void gemm_wmma_f16(
    const TA* __restrict__ A, int lda,
    const float* __restrict__ W,        // [N,K] row-major (i.e. B^T)
    const float* __restrict__ bias,     // [N] or nullptr
    float* __restrict__ C,              // [M,N] row-major
    int M, int N, int K, int act)       // act: 0=none, 1=softplus
{
  __shared__ _Float16 As[64][40];   // 40-half pitch = 80B (16B aligned rows)
  __shared__ _Float16 Bs[64][40];

  const int tid   = threadIdx.x;
  const int bm0   = blockIdx.x * 64;
  const int bn0   = blockIdx.y * 64;
  const int lane  = tid & 31;
  const int wid   = tid >> 5;
  const int wm    = wid >> 2;        // 0..1 : 32-row half
  const int wn    = wid & 3;         // 0..3 : 16-col group
  const int halfq = lane >> 4;       // 0/1
  const int mr    = lane & 15;       // M index (A frag), N index (B/D frag)

  v8f acc0 = {}, acc1 = {};

  for (int k0 = 0; k0 < K; k0 += 32) {
    // stage A tile 64x32: 512 4-elem chunks, 2 per thread
#pragma unroll
    for (int e = 0; e < 2; ++e) {
      int chunk = tid + e * 256;
      int r = chunk >> 3, c4 = (chunk & 7) * 4;
      *(h4*)&As[r][c4] = load4_h(&A[(size_t)(bm0 + r) * lda + (k0 + c4)]);
    }
    // stage W tile 64x32
#pragma unroll
    for (int e = 0; e < 2; ++e) {
      int chunk = tid + e * 256;
      int r = chunk >> 3, c4 = (chunk & 7) * 4;
      *(h4*)&Bs[r][c4] = load4_h(&W[(size_t)(bn0 + r) * K + (k0 + c4)]);
    }
    // prefetch next K tile while this one is consumed (L2-resident data)
    if (k0 + 32 < K) {
      __builtin_prefetch(&A[(size_t)(bm0 + (tid >> 2)) * lda + (k0 + 32)], 0, 0);
      __builtin_prefetch(&W[(size_t)(bn0 + (tid >> 2)) * K + (k0 + 32)], 0, 0);
    }
    __syncthreads();

    // A fragments for the two M tiles: lane m = mr;
    // VGPR i holds K pair at 2i + 8*half (+8 for i>=4)  [ISA 05_wmma layout]
    v16h a0, a1, bfrag;
#pragma unroll
    for (int i = 0; i < 8; ++i) {
      int kb = 2 * i + 8 * halfq + ((i >= 4) ? 8 : 0);
      a0[2 * i]     = As[wm * 32 + mr][kb];
      a0[2 * i + 1] = As[wm * 32 + mr][kb + 1];
      a1[2 * i]     = As[wm * 32 + 16 + mr][kb];
      a1[2 * i + 1] = As[wm * 32 + 16 + mr][kb + 1];
    }
    // B fragment: lane n = mr; same K order as A (8*half + j, +8 for j>=8)
#pragma unroll
    for (int j = 0; j < 16; ++j)
      bfrag[j] = Bs[wn * 16 + mr][(j & 7) + 8 * halfq + ((j >> 3) << 4)];

    acc0 = __builtin_amdgcn_wmma_f32_16x16x32_f16(
        false, a0, false, bfrag, (short)0, acc0, false, false);
    acc1 = __builtin_amdgcn_wmma_f32_16x16x32_f16(
        false, a1, false, bfrag, (short)0, acc1, false, false);
    __syncthreads();
  }

  // D layout: VGPR i -> row (i + 8*half), lane -> col
  const int col = bn0 + wn * 16 + mr;
  const float bv = bias ? bias[col] : 0.0f;
#pragma unroll
  for (int i = 0; i < 8; ++i) {
    int row0 = bm0 + wm * 32 + i + 8 * halfq;
    float v0 = acc0[i] + bv;
    if (act == 1) v0 = softplus_f(v0);
    C[(size_t)row0 * N + col] = v0;
    float v1 = acc1[i] + bv;
    if (act == 1) v1 = softplus_f(v1);
    C[(size_t)(row0 + 16) * N + col] = v1;
  }
}

// ---- conv stem -------------------------------------------------------------
// conv1 3->128, 3x3 pad1, + BN + GELU; writes NHWC h1[4,64,64,128]
__global__ void conv1_bn_gelu(const float* __restrict__ xin,
                              const float* __restrict__ w, const float* __restrict__ cb,
                              const float* __restrict__ g, const float* __restrict__ bb,
                              float* __restrict__ h1) {
  int idx = blockIdx.x * blockDim.x + threadIdx.x;       // 4*64*64*128
  int co = idx & 127, x = (idx >> 7) & 63, y = (idx >> 13) & 63, b = idx >> 19;
  float acc = cb[co];
#pragma unroll
  for (int ci = 0; ci < 3; ++ci)
#pragma unroll
    for (int ky = 0; ky < 3; ++ky)
#pragma unroll
      for (int kx = 0; kx < 3; ++kx) {
        int iy = y + ky - 1, ix = x + kx - 1;
        if ((unsigned)iy < 64u && (unsigned)ix < 64u)
          acc += xin[((b * 3 + ci) * 64 + iy) * 64 + ix] *
                 w[((co * 3 + ci) * 3 + ky) * 3 + kx];
      }
  float v = acc * (g[co] * rsqrtf(1.0f + EPSF)) + bb[co];
  h1[idx] = gelu_f(v);
}

// im2col for conv2 (3x3 pad1 over NHWC h1) -> f16 col[16384, 1152]
__global__ void im2col3x3(const float* __restrict__ h1, _Float16* __restrict__ col) {
  int idx = blockIdx.x * blockDim.x + threadIdx.x;       // 16384*1152
  int c = idx % 1152, r = idx / 1152;
  int ch = c / 9, kk = c - ch * 9, ky = kk / 3, kx = kk - ky * 3;
  int b = r >> 12, y = (r >> 6) & 63, x = r & 63;
  int iy = y + ky - 1, ix = x + kx - 1;
  float v = ((unsigned)iy < 64u && (unsigned)ix < 64u)
                ? h1[((b * 64 + iy) * 64 + ix) * 128 + ch] : 0.0f;
  col[idx] = (_Float16)v;
}

// in-place BN + GELU over NHWC [*, C] activations
__global__ void bn_gelu_inplace(float* __restrict__ h, const float* __restrict__ g,
                                const float* __restrict__ bb, int C) {
  int idx = blockIdx.x * blockDim.x + threadIdx.x;
  int ch = idx & (C - 1);
  float v = h[idx] * (g[ch] * rsqrtf(1.0f + EPSF)) + bb[ch];
  h[idx] = gelu_f(v);
}

// patch conv 2x2 stride2 + BN + GELU, NHWC h2 -> sequence s[4,1024,256]
__global__ void patch_bn_gelu(const float* __restrict__ h2,
                              const float* __restrict__ w, const float* __restrict__ pb,
                              const float* __restrict__ g, const float* __restrict__ bb,
                              float* __restrict__ s) {
  int idx = blockIdx.x * blockDim.x + threadIdx.x;       // 4*32*32*256
  int n = idx & 255, ox = (idx >> 8) & 31, oy = (idx >> 13) & 31, b = idx >> 18;
  float acc = pb[n];
#pragma unroll
  for (int ky = 0; ky < 2; ++ky)
#pragma unroll
    for (int kx = 0; kx < 2; ++kx) {
      const float* hp = h2 + (size_t)((b * 64 + 2 * oy + ky) * 64 + (2 * ox + kx)) * 256;
      const float* wp = w + (size_t)n * 1024 + ky * 2 + kx;
      for (int c = 0; c < 256; ++c) acc += hp[c] * wp[c * 4];
    }
  float v = acc * (g[n] * rsqrtf(1.0f + EPSF)) + bb[n];
  s[(size_t)(b * 1024 + oy * 32 + ox) * 256 + n] = gelu_f(v);
}

// ---- mamba pieces ----------------------------------------------------------
// depthwise conv1d over L, kernel 4, pad (1,2); input = xi half of xr
__global__ void dwconv1d(const float* __restrict__ xr, const float* __restrict__ cw,
                         const float* __restrict__ cb, float* __restrict__ xc) {
  int idx = blockIdx.x * blockDim.x + threadIdx.x;       // 4096*512
  int c = idx & (DI - 1);
  int tok = idx >> 9;
  int l = tok & (LSEQ - 1), b = tok >> 10;
  float acc = cb[c];
#pragma unroll
  for (int k = 0; k < 4; ++k) {
    int ls = l + k - 1;
    if ((unsigned)ls < (unsigned)LSEQ)
      acc += cw[c * 4 + k] * xr[(size_t)((b << 10) + ls) * (2 * DI) + c];
  }
  xc[idx] = acc;
}

// selective scan + fused SiLU gating; one thread per (b, d)
__global__ void selective_scan(const float* __restrict__ u,      // xc [4096,512]
                               const float* __restrict__ delta,  // [4096,512]
                               const float* __restrict__ xdbl,   // [4096,64] (B at +32, C at +48)
                               const float* __restrict__ alog,   // [512,16]
                               const float* __restrict__ Dp,     // [512]
                               const float* __restrict__ xr,     // res = xr[:, 512:1024]
                               float* __restrict__ y) {
  int t = blockIdx.x * blockDim.x + threadIdx.x;
  if (t >= MB * DI) return;
  int b = t >> 9, d = t & (DI - 1);
  float Av[NS], h[NS];
#pragma unroll
  for (int n = 0; n < NS; ++n) { Av[n] = -expf(alog[d * NS + n]); h[n] = 0.0f; }
  float Dd = Dp[d];
  for (int l = 0; l < LSEQ; ++l) {
    int tok = (b << 10) + l;
    float uv = u[(size_t)tok * DI + d];
    float dv = delta[(size_t)tok * DI + d];
    const float* bc = xdbl + (size_t)tok * 64;
    float acc = 0.0f;
#pragma unroll
    for (int n = 0; n < NS; ++n) {
      h[n] = expf(dv * Av[n]) * h[n] + dv * bc[32 + n] * uv;
      acc += h[n] * bc[48 + n];
    }
    float yy = acc + uv * Dd;
    float r = xr[(size_t)tok * (2 * DI) + DI + d];
    y[(size_t)tok * DI + d] = yy * silu_f(r);
  }
}

// LN over channels (256) + residual add into s; one block per token
__global__ void ln_residual(const float* __restrict__ yin, const float* __restrict__ g,
                            const float* __restrict__ bb, float* __restrict__ s) {
  __shared__ float red[256];
  int tok = blockIdx.x, c = threadIdx.x;
  float v = yin[(size_t)tok * DM + c];
  float mean = block_reduce_sum256(v, red) * (1.0f / DM);
  float d = v - mean;
  float var = block_reduce_sum256(d * d, red) * (1.0f / DM);
  s[(size_t)tok * DM + c] += d * rsqrtf(var + EPSF) * g[c] + bb[c];
}

// ---- head ------------------------------------------------------------------
__global__ void mean_pool(const float* __restrict__ s, float* __restrict__ pooled) {
  int t = blockIdx.x * blockDim.x + threadIdx.x;         // 4*256
  int b = t >> 8, c = t & 255;
  float acc = 0.0f;
  for (int l = 0; l < LSEQ; ++l) acc += s[(size_t)((b << 10) + l) * DM + c];
  pooled[b * DM + c] = acc * (1.0f / LSEQ);
}

__global__ void ln_head(const float* __restrict__ pooled, const float* __restrict__ g,
                        const float* __restrict__ bb, float* __restrict__ out) {
  __shared__ float red[256];
  int b = blockIdx.x, c = threadIdx.x;
  float v = pooled[b * DM + c];
  float mean = block_reduce_sum256(v, red) * (1.0f / DM);
  float d = v - mean;
  float var = block_reduce_sum256(d * d, red) * (1.0f / DM);
  out[b * DM + c] = d * rsqrtf(var + EPSF) * g[c] + bb[c];
}

__global__ void fc_head(const float* __restrict__ x, const float* __restrict__ w,
                        const float* __restrict__ bb, float* __restrict__ logits) {
  int t = blockIdx.x * blockDim.x + threadIdx.x;
  if (t >= MB * NC) return;
  int b = t / NC, n = t - b * NC;
  float acc = bb[n];
  for (int k = 0; k < DM; ++k) acc += x[b * DM + k] * w[n * DM + k];
  logits[t] = acc;
}

__global__ void softmax_head(const float* __restrict__ logits, float* __restrict__ out) {
  __shared__ float red[256];
  int b = blockIdx.x, tid = threadIdx.x;
  float m = -3.4e38f;
  for (int i = tid; i < NC; i += 256) m = fmaxf(m, logits[b * NC + i]);
  red[tid] = m; __syncthreads();
  for (int s = 128; s > 0; s >>= 1) { if (tid < s) red[tid] = fmaxf(red[tid], red[tid + s]); __syncthreads(); }
  m = red[0]; __syncthreads();
  float sum = 0.0f;
  for (int i = tid; i < NC; i += 256) sum += expf(logits[b * NC + i] - m);
  red[tid] = sum; __syncthreads();
  for (int s = 128; s > 0; s >>= 1) { if (tid < s) red[tid] += red[tid + s]; __syncthreads(); }
  float inv = 1.0f / red[0];
  for (int i = tid; i < NC; i += 256) out[b * NC + i] = expf(logits[b * NC + i] - m) * inv;
}

// ---- host orchestration ----------------------------------------------------
extern "C" void kernel_launch(void* const* d_in, const int* in_sizes, int n_in,
                              void* d_out, int out_size, void* d_ws, size_t ws_size,
                              hipStream_t stream) {
  // inputs (setup_inputs order)
  const float* x    = (const float*)d_in[0];
  const float* c1w  = (const float*)d_in[1];
  const float* c1b  = (const float*)d_in[2];
  const float* g1   = (const float*)d_in[3];
  const float* b1   = (const float*)d_in[4];
  const float* c2w  = (const float*)d_in[5];
  const float* c2b  = (const float*)d_in[6];
  const float* g2   = (const float*)d_in[7];
  const float* b2   = (const float*)d_in[8];
  const float* pw   = (const float*)d_in[9];
  const float* pb   = (const float*)d_in[10];
  const float* g3   = (const float*)d_in[11];
  const float* b3   = (const float*)d_in[12];
  const float* ipw  = (const float*)d_in[13];
  const float* ipb  = (const float*)d_in[14];
  const float* cw   = (const float*)d_in[15];
  const float* cb   = (const float*)d_in[16];
  const float* xpw  = (const float*)d_in[17];
  const float* dpw  = (const float*)d_in[18];
  const float* dpb  = (const float*)d_in[19];
  const float* alog = (const float*)d_in[20];
  const float* dssm = (const float*)d_in[21];
  const float* opw  = (const float*)d_in[22];
  const float* opb  = (const float*)d_in[23];
  const float* lnw  = (const float*)d_in[24];
  const float* lnb  = (const float*)d_in[25];
  const float* nw   = (const float*)d_in[26];
  const float* nb   = (const float*)d_in[27];
  const float* fcw  = (const float*)d_in[28];
  const float* fcb  = (const float*)d_in[29];

  char* ws = (char*)d_ws;
  float*     s_buf  = (float*)(ws + OFF_S);
  float*     h1     = (float*)(ws + OFF_H1);
  float*     h2     = (float*)(ws + OFF_H2);
  _Float16*  col    = (_Float16*)(ws + OFF_COL);
  float*     xr     = (float*)(ws + OFF_XR);
  float*     xc     = (float*)(ws + OFF_XC);
  float*     xdbl   = (float*)(ws + OFF_XDBL);
  float*     delta  = (float*)(ws + OFF_DELTA);
  float*     ybuf   = (float*)(ws + OFF_Y);
  float*     outb   = (float*)(ws + OFF_OUT);
  float*     pooled = (float*)(ws + OFF_POOL);
  float*     pooln  = (float*)(ws + OFF_POOLN);

  float* logits = (float*)d_out;          // [4,1000]
  float* smax   = logits + MB * NC;       // [4,1000]

  // ---- stem ----
  conv1_bn_gelu<<<(4 * 64 * 64 * 128) / 256, 256, 0, stream>>>(x, c1w, c1b, g1, b1, h1);
  im2col3x3<<<(16384 * 1152) / 256, 256, 0, stream>>>(h1, col);
  gemm_wmma_f16<_Float16><<<dim3(16384 / 64, 256 / 64), 256, 0, stream>>>(
      col, 1152, c2w, c2b, h2, 16384, 256, 1152, 0);
  bn_gelu_inplace<<<(16384 * 256) / 256, 256, 0, stream>>>(h2, g2, b2, 256);
  patch_bn_gelu<<<(4 * 32 * 32 * 256) / 256, 256, 0, stream>>>(h2, pw, pb, g3, b3, s_buf);

  // ---- mamba layers ----
  for (int i = 0; i < NL; ++i) {
    const float* ipw_i = ipw + (size_t)i * 2 * DI * DM;
    const float* ipb_i = ipb + (size_t)i * 2 * DI;
    const float* cw_i  = cw  + (size_t)i * DI * 4;
    const float* cb_i  = cb  + (size_t)i * DI;
    const float* xpw_i = xpw + (size_t)i * (DTR + 2 * NS) * DI;
    const float* dpw_i = dpw + (size_t)i * DI * DTR;
    const float* dpb_i = dpb + (size_t)i * DI;
    const float* al_i  = alog+ (size_t)i * DI * NS;
    const float* ds_i  = dssm+ (size_t)i * DI;
    const float* opw_i = opw + (size_t)i * DM * DI;
    const float* opb_i = opb + (size_t)i * DM;
    const float* lnw_i = lnw + (size_t)i * DM;
    const float* lnb_i = lnb + (size_t)i * DM;

    // in_proj: [4096,256] x [1024,256]^T -> xr [4096,1024]
    gemm_wmma_f16<float><<<dim3(NTOK / 64, (2 * DI) / 64), 256, 0, stream>>>(
        s_buf, DM, ipw_i, ipb_i, xr, NTOK, 2 * DI, DM, 0);
    // depthwise conv1d on xi half
    dwconv1d<<<(NTOK * DI) / 256, 256, 0, stream>>>(xr, cw_i, cb_i, xc);
    // x_proj: [4096,512] x [64,512]^T -> xdbl [4096,64] (no bias)
    gemm_wmma_f16<float><<<dim3(NTOK / 64, 64 / 64), 256, 0, stream>>>(
        xc, DI, xpw_i, nullptr, xdbl, NTOK, 64, DI, 0);
    // dt_proj + softplus: view xdbl[:, :32] (lda=64) x [512,32]^T -> delta
    gemm_wmma_f16<float><<<dim3(NTOK / 64, DI / 64), 256, 0, stream>>>(
        xdbl, 64, dpw_i, dpb_i, delta, NTOK, DI, DTR, 1);
    // selective scan + SiLU gating
    selective_scan<<<(MB * DI) / 256, 256, 0, stream>>>(
        xc, delta, xdbl, al_i, ds_i, xr, ybuf);
    // out_proj: [4096,512] x [256,512]^T -> outb [4096,256]
    gemm_wmma_f16<float><<<dim3(NTOK / 64, DM / 64), 256, 0, stream>>>(
        ybuf, DI, opw_i, opb_i, outb, NTOK, DM, DI, 0);
    // LN + residual into s
    ln_residual<<<NTOK, 256, 0, stream>>>(outb, lnw_i, lnb_i, s_buf);
  }

  // ---- head ----
  mean_pool<<<(MB * DM) / 256, 256, 0, stream>>>(s_buf, pooled);
  ln_head<<<MB, 256, 0, stream>>>(pooled, nw, nb, pooln);
  fc_head<<<(MB * NC + 255) / 256, 256, 0, stream>>>(pooln, fcw, fcb, logits);
  softmax_head<<<MB, 256, 0, stream>>>(logits, smax);
}
